// WPLICalculator_9423158248173
// MI455X (gfx1250) — matrix operand order repaired and checked by reading, not compiled, not run on hardware
//
#include <hip/hip_runtime.h>
#include <hip/hip_bf16.h>

#define Bsz 8
#define Csz 64
#define Tsz 2048
#define NSIG (Bsz * Csz)   // 512 signals

typedef __attribute__((ext_vector_type(16))) _Float16 v16h;
typedef __attribute__((ext_vector_type(8)))  _Float16 v8h;
typedef __attribute__((ext_vector_type(8)))  float    v8f;

// ---------------------------------------------------------------------------
// Kernel 1: prep. Convert x -> f16 (WMMA A operand) and compute per-row
// parity sums for the exact real part of the analytic signal:
//   re[t] = x[t] + (2/N) * S_{opposite parity of t}
// ---------------------------------------------------------------------------
__global__ void __launch_bounds__(256)
wpli_prep(const float* __restrict__ x, _Float16* __restrict__ Xh,
          float* __restrict__ addc) {
  __shared__ float red[256];
  const int row = blockIdx.x;
  const int tid = threadIdx.x;
  const float* xr = x + (size_t)row * Tsz;
  float spar = 0.f;                       // parity(idx) == parity(tid)
  for (int k = 0; k < Tsz / 256; ++k) {
    int idx = tid + 256 * k;
    float v = xr[idx];
    Xh[(size_t)row * Tsz + idx] = (_Float16)v;
    spar += v;
  }
  red[tid] = spar;
  __syncthreads();
  if (tid < 2) {                          // tid 0 -> S_even, tid 1 -> S_odd
    float acc = 0.f;
    for (int i2 = tid; i2 < 256; i2 += 2) acc += red[i2];
    // even t adds (2/N)*S_odd (slot 0), odd t adds (2/N)*S_even (slot 1)
    addc[row * 2 + (tid ^ 1)] = (2.0f / (float)Tsz) * acc;
  }
}

// ---------------------------------------------------------------------------
// Kernel 2: circulant Hilbert operand (closed form of ifft(fft(.)*(k<N/2)*2)):
//   g[d] = (2/N)*cot(pi*d/N) for odd d, else 0.
// GhT[t][s] = g[(t-s) mod N], row-major in t so the WMMA B fragment
// (K-run of 16 at fixed column t) is a contiguous 32B load.
// ---------------------------------------------------------------------------
__global__ void __launch_bounds__(256)
wpli_build_g(_Float16* __restrict__ GhT) {
  __shared__ float gl[Tsz];
  const int t = blockIdx.x;
  const int tid = threadIdx.x;
  const float twoOverN = 2.0f / (float)Tsz;
  const float PI = 3.14159265358979323846f;
  for (int j = tid; j < Tsz / 2; j += 256) {
    float a = PI * (float)(2 * j + 1) / (float)Tsz;
    gl[2 * j]     = 0.f;
    gl[2 * j + 1] = twoOverN * (cosf(a) / sinf(a));
  }
  __syncthreads();
  _Float16* grow = GhT + (size_t)t * Tsz;
  for (int s = tid; s < Tsz; s += 256)
    grow[s] = (_Float16)gl[(t - s) & (Tsz - 1)];
}

// ---------------------------------------------------------------------------
// Fused phase epilogue for one 16x16 D tile (8 rows per lane):
// re from f32 x + parity constant, normalize to unit phasor, store f16.
// ---------------------------------------------------------------------------
__device__ __forceinline__ void
wpli_phase_epilogue(const v8f& acc, int row0, int t,
                    const float* __restrict__ x,
                    const float* __restrict__ addc,
                    _Float16* __restrict__ Uh, _Float16* __restrict__ Vh) {
  const int par = t & 1;
#pragma unroll
  for (int rr = 0; rr < 8; ++rr) {
    const int row = row0 + rr;
    float im = acc[rr];
    float re = x[(size_t)row * Tsz + t] + addc[row * 2 + par];
    float mag = sqrtf(re * re + im * im);
    float inv = (mag > 0.f) ? 1.f / mag : 0.f;
    float u = (mag > 0.f) ? re * inv : 1.f;   // angle(0)==0 -> (1,0)
    float v = im * inv;
    Uh[(size_t)row * Tsz + t] = (_Float16)u;
    Vh[(size_t)row * Tsz + t] = (_Float16)v;
  }
}

#define COMBINE16(lo, hi) \
  __builtin_shufflevector(lo, hi, 0,1,2,3,4,5,6,7,8,9,10,11,12,13,14,15)

// ---------------------------------------------------------------------------
// Kernel 3: Hilbert via WMMA GEMM, Im = Xh (512x2048) @ GhT^T.
// 2x2 register blocking: one wave owns a 32x32 output region (4 tiles,
// 4 f32 accumulators).  Per K-step of 32: 8 x b128 loads feed 4 WMMAs
// (2:1 vmem:wmma vs 4:1 unblocked), halving L2 traffic and VMEM issue.
// A lane layout (16x32 f16): lane<16 row=lane, K-run {0..15};
//                            lane>=16 row=lane-16, K-run {16..31}.
// B lane layout (32x16 f16): col = lane&15, K-run 16 at (lane>=16 ? 16 : 0).
// ---------------------------------------------------------------------------
__global__ void __launch_bounds__(256)
wpli_hilbert_wmma(const float* __restrict__ x,
                  const _Float16* __restrict__ Xh,
                  const _Float16* __restrict__ GhT,
                  const float* __restrict__ addc,
                  _Float16* __restrict__ Uh,
                  _Float16* __restrict__ Vh) {
  const int lane = threadIdx.x & 31;
  const int waveIdx = blockIdx.x * 8 + (threadIdx.x >> 5);  // 0..1023
  const int bm = waveIdx >> 6;     // 0..15  (M blocks of 32 rows)
  const int bn = waveIdx & 63;     // 0..63  (N blocks of 32 cols)
  const int hs  = (lane >> 4) & 1;
  const int l15 = lane & 15;

  const _Float16* Arow0 = Xh  + ((size_t)(bm * 32 + l15)) * Tsz;
  const _Float16* Arow1 = Arow0 + (size_t)16 * Tsz;
  const _Float16* Brow0 = GhT + ((size_t)(bn * 32 + l15)) * Tsz;
  const _Float16* Brow1 = Brow0 + (size_t)16 * Tsz;
  const int ca = hs * 16;
  const int cb = hs * 16;

  v8f acc00 = {}, acc01 = {}, acc10 = {}, acc11 = {};
  for (int k0 = 0; k0 < Tsz; k0 += 32) {
    v8h a0l = *(const v8h*)(Arow0 + k0 + ca);
    v8h a0h = *(const v8h*)(Arow0 + k0 + ca + 8);
    v8h a1l = *(const v8h*)(Arow1 + k0 + ca);
    v8h a1h = *(const v8h*)(Arow1 + k0 + ca + 8);
    v8h b0l = *(const v8h*)(Brow0 + k0 + cb);
    v8h b0h = *(const v8h*)(Brow0 + k0 + cb + 8);
    v8h b1l = *(const v8h*)(Brow1 + k0 + cb);
    v8h b1h = *(const v8h*)(Brow1 + k0 + cb + 8);
    v16h A0 = COMBINE16(a0l, a0h);
    v16h A1 = COMBINE16(a1l, a1h);
    v16h B0 = COMBINE16(b0l, b0h);
    v16h B1 = COMBINE16(b1l, b1h);
    acc00 = __builtin_amdgcn_wmma_f32_16x16x32_f16(false, A0, false, B0, (short)0, acc00, false, false);
    acc01 = __builtin_amdgcn_wmma_f32_16x16x32_f16(false, A0, false, B1, (short)0, acc01, false, false);
    acc10 = __builtin_amdgcn_wmma_f32_16x16x32_f16(false, A1, false, B0, (short)0, acc10, false, false);
    acc11 = __builtin_amdgcn_wmma_f32_16x16x32_f16(false, A1, false, B1, (short)0, acc11, false, false);
  }

  const int row00 = bm * 32 + hs * 8;         // D: VGPR rr -> M = rr + hs*8
  const int t0    = bn * 32 + l15;
  wpli_phase_epilogue(acc00, row00,      t0,      x, addc, Uh, Vh);
  wpli_phase_epilogue(acc01, row00,      t0 + 16, x, addc, Uh, Vh);
  wpli_phase_epilogue(acc10, row00 + 16, t0,      x, addc, Uh, Vh);
  wpli_phase_epilogue(acc11, row00 + 16, t0 + 16, x, addc, Uh, Vh);
}

// ---------------------------------------------------------------------------
// Kernel 4: P[b] = V[b] * U[b]^T  (M=N=64, K=2048) via WMMA. One wave per
// 16x16 tile: 8 batches * 16 tiles = 128 waves = 16 blocks.
// numerator = |P - P^T| / T  (sin(pi-pj) = vi*uj - ui*vj summed over t).
// ---------------------------------------------------------------------------
__global__ void __launch_bounds__(256)
wpli_numerator_wmma(const _Float16* __restrict__ Uh,
                    const _Float16* __restrict__ Vh,
                    float* __restrict__ P) {
  const int lane = threadIdx.x & 31;
  const int wave = blockIdx.x * 8 + (threadIdx.x >> 5);
  const int b    = wave >> 4;
  const int tile = wave & 15;
  const int ti = tile >> 2, tj = tile & 3;
  const int hs  = (lane >> 4) & 1;
  const int l15 = lane & 15;

  const _Float16* Arow = Vh + ((size_t)(b * Csz + ti * 16 + l15)) * Tsz;
  const _Float16* Brow = Uh + ((size_t)(b * Csz + tj * 16 + l15)) * Tsz;
  const int ca = hs * 16;
  const int cb = hs * 16;

  v8f acc = {};
#pragma unroll 2
  for (int k0 = 0; k0 < Tsz; k0 += 32) {
    v8h a0 = *(const v8h*)(Arow + k0 + ca);
    v8h a1 = *(const v8h*)(Arow + k0 + ca + 8);
    v8h b0 = *(const v8h*)(Brow + k0 + cb);
    v8h b1 = *(const v8h*)(Brow + k0 + cb + 8);
    v16h A  = COMBINE16(a0, a1);
    v16h Bm = COMBINE16(b0, b1);
    acc = __builtin_amdgcn_wmma_f32_16x16x32_f16(
        false, A, false, Bm, (short)0, acc, false, false);
  }

  float* Pb = P + (size_t)b * Csz * Csz;
  const int Mb = hs * 8;
#pragma unroll
  for (int r = 0; r < 8; ++r)
    Pb[(ti * 16 + Mb + r) * Csz + tj * 16 + l15] = acc[r];
}

// ---------------------------------------------------------------------------
// Kernel 5: denominator (mean_t |vi*uj - ui*vj|; abs blocks matmul form)
// + final WPLI.  One block per (b,i); row-i phasors in LDS; 8 waves each
// reduce their own j columns with wave32 shuffle reduction.
// ---------------------------------------------------------------------------
__global__ void __launch_bounds__(256)
wpli_denom_final(const _Float16* __restrict__ Uh,
                 const _Float16* __restrict__ Vh,
                 const float* __restrict__ P,
                 float* __restrict__ out) {
  __shared__ float ui[Tsz];
  __shared__ float vi[Tsz];
  const int b = blockIdx.x >> 6;
  const int i = blockIdx.x & 63;
  const int tid = threadIdx.x;

  const _Float16* Ui = Uh + ((size_t)(b * Csz + i)) * Tsz;
  const _Float16* Vi = Vh + ((size_t)(b * Csz + i)) * Tsz;
  for (int t = tid; t < Tsz; t += 256) {
    ui[t] = (float)Ui[t];
    vi[t] = (float)Vi[t];
  }
  __syncthreads();

  const int lane = tid & 31;
  const int wv   = tid >> 5;
  const float invT = 1.0f / (float)Tsz;

  for (int jj = 0; jj < 8; ++jj) {
    const int j = wv + jj * 8;
    const _Float16* Uj = Uh + ((size_t)(b * Csz + j)) * Tsz;
    const _Float16* Vj = Vh + ((size_t)(b * Csz + j)) * Tsz;
    float part = 0.f;
    for (int t = lane; t < Tsz; t += 32) {
      float uj = (float)Uj[t];
      float vj = (float)Vj[t];
      part += fabsf(vi[t] * uj - ui[t] * vj);
    }
    for (int off = 16; off > 0; off >>= 1)      // wave32 reduction
      part += __shfl_down(part, off, 32);
    if (lane == 0) {
      float den = part * invT;
      float num = fabsf(P[((size_t)b * Csz + i) * Csz + j] -
                        P[((size_t)b * Csz + j) * Csz + i]) * invT;
      if (den == 0.f) den = 1e-8f;              // faithful to reference
      float val = (i == j) ? 0.f : num / den;   // zero diagonal
      out[((size_t)b * Csz + i) * Csz + j] = val;
    }
  }
}

// ---------------------------------------------------------------------------
extern "C" void kernel_launch(void* const* d_in, const int* in_sizes, int n_in,
                              void* d_out, int out_size, void* d_ws, size_t ws_size,
                              hipStream_t stream) {
  (void)in_sizes; (void)n_in; (void)out_size; (void)ws_size;
  const float* x = (const float*)d_in[0];
  char* ws = (char*)d_ws;
  const size_t szSig = (size_t)NSIG * Tsz;          // 1 Mi elements
  _Float16* Xh  = (_Float16*)ws;                                  // 2 MB
  _Float16* Uh  = (_Float16*)(ws + szSig * 2);                    // 2 MB
  _Float16* Vh  = (_Float16*)(ws + szSig * 4);                    // 2 MB
  _Float16* GhT = (_Float16*)(ws + szSig * 6);                    // 8 MB
  float*    P   = (float*)   (ws + szSig * 6 + (size_t)Tsz * Tsz * 2);       // 128 KB
  float*    addc= (float*)   (ws + szSig * 6 + (size_t)Tsz * Tsz * 2 + (size_t)Bsz * Csz * Csz * 4); // 4 KB
  float* out = (float*)d_out;

  wpli_prep          <<<NSIG, 256, 0, stream>>>(x, Xh, addc);
  wpli_build_g       <<<Tsz,  256, 0, stream>>>(GhT);
  wpli_hilbert_wmma  <<<(NSIG/32)*(Tsz/32)/8, 256, 0, stream>>>(x, Xh, GhT, addc, Uh, Vh);
  wpli_numerator_wmma<<<(Bsz * 16) / 8, 256, 0, stream>>>(Uh, Vh, P);
  wpli_denom_final   <<<NSIG, 256, 0, stream>>>(Uh, Vh, P, out);
}
